// LSTMCell_35450660061928
// MI455X (gfx1250) — compile-verified, matrix operand order checked by reading, not run on hardware
//
#include <hip/hip_runtime.h>

typedef __attribute__((ext_vector_type(16))) _Float16 v16h;
typedef __attribute__((ext_vector_type(8)))  _Float16 v8h;
typedef __attribute__((ext_vector_type(2)))  _Float16 v2h;
typedef __attribute__((ext_vector_type(8)))  float    v8f;

#define BSZ  4096
#define INSZ 1024
#define HID  1024
#define KDIM 2048   // INSZ + HID

#define KC    64                 // K-chunk staged in LDS
#define BROW  72                 // padded LDS row stride in halfs (64 + 8 pad)
#define BBUF  (4 * 32 * BROW)    // halfs per LDS buffer (4 gates x 32 cols)
#define NKC   (KDIM / KC)        // 32 chunks

// ---------------------------------------------------------------------------
// Kernel 1: combined = f16(cat([x, h_prev], axis=1))   [BSZ x KDIM] f16
// ---------------------------------------------------------------------------
__global__ __launch_bounds__(256) void build_combined(
    const float* __restrict__ x, const float* __restrict__ h,
    _Float16* __restrict__ comb) {
  int idx = blockIdx.x * blockDim.x + threadIdx.x;   // over BSZ*KDIM/2 pairs
  int row = idx >> 10;                               // KDIM/2 = 1024 pairs/row
  int p   = idx & 1023;
  int c   = p * 2;
  const float* src = (c < INSZ) ? (x + (size_t)row * INSZ + c)
                                : (h + (size_t)row * HID + (c - INSZ));
  float2 v = *(const float2*)src;
  v2h o;
  o.x = (_Float16)v.x;
  o.y = (_Float16)v.y;
  *(v2h*)(comb + (size_t)row * KDIM + c) = o;
}

// ---------------------------------------------------------------------------
// Kernel 2: Wt[g][n][k] = f16(W_g[k][n])  (LDS-tiled transpose, coalesced)
// ---------------------------------------------------------------------------
__global__ __launch_bounds__(256) void wt_kernel(
    const float* __restrict__ Wf, const float* __restrict__ Wi,
    const float* __restrict__ Wc, const float* __restrict__ Wo,
    _Float16* __restrict__ Wt) {
  __shared__ float tile[32][33];
  const float* Wsel[4] = {Wf, Wi, Wc, Wo};
  const int g = blockIdx.z;
  const float* __restrict__ src = Wsel[g];
  _Float16* __restrict__ dst = Wt + (size_t)g * HID * KDIM;

  const int nbase = blockIdx.x * 32;
  const int kbase = blockIdx.y * 32;
  const int tx = threadIdx.x;   // 0..31
  const int ty = threadIdx.y;   // 0..7

#pragma unroll
  for (int j = 0; j < 32; j += 8)
    tile[ty + j][tx] = src[(size_t)(kbase + ty + j) * HID + nbase + tx];
  __syncthreads();
#pragma unroll
  for (int j = 0; j < 32; j += 8)
    dst[(size_t)(nbase + ty + j) * KDIM + kbase + tx] =
        (_Float16)tile[tx][ty + j];
}

// ---------------------------------------------------------------------------
// Kernel 3: fused 4-gate WMMA GEMM + LSTM pointwise, LDS-staged B panels.
// Block = 8 waves: 128 M x 32 N x 4 gates. B panels double-buffered in LDS
// in K-chunks of 64; A direct from L2. Biases pre-loaded into accumulators.
// Grid: (BSZ/128) * (HID/32) = 1024 blocks.
// ---------------------------------------------------------------------------
__device__ __forceinline__ float sigf(float v) {
  return 1.0f / (1.0f + __expf(-v));
}

__global__ __launch_bounds__(256) void lstm_gemm(
    const _Float16* __restrict__ comb, const _Float16* __restrict__ Wt,
    const float* __restrict__ c_prev,
    const float* __restrict__ bf, const float* __restrict__ bi,
    const float* __restrict__ bc, const float* __restrict__ bo,
    float* __restrict__ out) {
  __shared__ __align__(16) _Float16 ldsB[2][BBUF];

  const int tid   = threadIdx.x;
  const int wave  = tid >> 5;
  const int lane  = tid & 31;
  const int blkN  = blockIdx.x & 31;         // HID/32 = 32
  const int blkM  = blockIdx.x >> 5;         // BSZ/128 = 32
  const int m0    = blkM * 128 + wave * 16;
  const int n0    = blkN * 32;

  const int laneM  = lane & 15;
  const int laneHi = lane >> 4;              // 0 or 1
  const int koff   = laneHi << 3;            // 0 or 8 (fragment K offset)

  // --- cooperative B-fill descriptors: 4 x b128 (8 halfs) per thread/chunk
  const _Float16* gptr[4];
  int loff[4];
#pragma unroll
  for (int j = 0; j < 4; ++j) {
    const int id  = tid + 256 * j;           // 0..1023
    const int kk8 = id & 7;                  // 8 chunks per 64-K row
    const int c   = (id >> 3) & 31;
    const int g   = id >> 8;
    gptr[j] = Wt + ((size_t)g * HID + (size_t)(n0 + c)) * KDIM + kk8 * 8;
    loff[j] = (g * 32 + c) * BROW + kk8 * 8;
  }

  // A fragment source: row m0+laneM, K = 8*laneHi+s, then 16+8*laneHi+(s-8) (ISA 16-bit A layout)
  const _Float16* __restrict__ arow = comb + (size_t)(m0 + laneM) * KDIM + koff;

  // LDS read offsets per (gate, ntile): row = g*32 + t*16 + laneM
  int rowB[4][2];
#pragma unroll
  for (int g = 0; g < 4; ++g)
#pragma unroll
    for (int t = 0; t < 2; ++t)
      rowB[g][t] = (g * 32 + t * 16 + laneM) * BROW + koff;

  // Accumulators initialized with the broadcast bias: C/D layout puts one
  // column per lane (col = lane&15), all 8 VGPR rows share that column.
  v8f acc[4][2];
#pragma unroll
  for (int t = 0; t < 2; ++t) {
    const int col = n0 + t * 16 + laneM;
    acc[0][t] = (v8f)bf[col];
    acc[1][t] = (v8f)bi[col];
    acc[2][t] = (v8f)bc[col];
    acc[3][t] = (v8f)bo[col];
  }

  // --- prologue: fill buffer 0 with chunk 0
#pragma unroll
  for (int j = 0; j < 4; ++j)
    *(v8h*)(&ldsB[0][0] + loff[j]) = *(const v8h*)(gptr[j]);

  int cur = 0;
  for (int kc = 0; kc < NKC; ++kc) {
    __syncthreads();  // buf[cur] visible; prior reads of buf[cur^1] done

    const int kg = kc * KC;

    // A loads first (clause), then next-chunk stage loads. Loads complete in
    // order, so the first WMMA only needs a partial loadcnt wait for A.
    v8h a0lo = *(const v8h*)(arow + kg);
    v8h a0hi = *(const v8h*)(arow + kg + 16);
    v8h a1lo = *(const v8h*)(arow + kg + 32);
    v8h a1hi = *(const v8h*)(arow + kg + 48);

    v8h stage[4];
    const bool more = (kc + 1) < NKC;
    if (more) {
#pragma unroll
      for (int j = 0; j < 4; ++j)
        stage[j] = *(const v8h*)(gptr[j] + (kc + 1) * KC);
    }
#if __has_builtin(__builtin_amdgcn_sched_barrier)
    __builtin_amdgcn_sched_barrier(0);  // keep stage loads issued before WMMAs
#endif

    v16h a[2];
    a[0] = __builtin_shufflevector(a0lo, a0hi, 0, 1, 2, 3, 4, 5, 6, 7, 8, 9,
                                   10, 11, 12, 13, 14, 15);
    a[1] = __builtin_shufflevector(a1lo, a1hi, 0, 1, 2, 3, 4, 5, 6, 7, 8, 9,
                                   10, 11, 12, 13, 14, 15);

    const _Float16* bp = &ldsB[cur][0];
#pragma unroll
    for (int ks = 0; ks < 2; ++ks) {
      // Load all 8 B fragments into distinct registers first so the backend
      // issues all 16 ds_loads and then uses partial s_wait_dscnt per WMMA.
      v16h bfr[8];
#pragma unroll
      for (int g = 0; g < 4; ++g) {
#pragma unroll
        for (int t = 0; t < 2; ++t) {
          const int off = rowB[g][t] + ks * 32;
          v8h blo = *(const v8h*)(bp + off);
          v8h bhi = *(const v8h*)(bp + off + 16);
          bfr[g * 2 + t] = __builtin_shufflevector(
              blo, bhi, 0, 1, 2, 3, 4, 5, 6, 7, 8, 9, 10, 11, 12, 13, 14, 15);
        }
      }
#pragma unroll
      for (int g = 0; g < 4; ++g) {
#pragma unroll
        for (int t = 0; t < 2; ++t) {
          acc[g][t] = __builtin_amdgcn_wmma_f32_16x16x32_f16(
              false, a[ks], false, bfr[g * 2 + t], (short)0, acc[g][t], false,
              false);
        }
      }
#if __has_builtin(__builtin_amdgcn_sched_group_barrier)
      // Enforce the shape: [16 x DS-read][8 x WMMA] per k-step, so the DS
      // pipe runs ahead of the matrix pipe with partial dscnt waits.
      __builtin_amdgcn_sched_group_barrier(0x100, 16, 0);  // DS reads
      __builtin_amdgcn_sched_group_barrier(0x008, 8, 0);   // WMMA/MFMA
#endif
    }

    // Store staged chunk into the other buffer (stage loads long in flight)
    if (more) {
      _Float16* wp = &ldsB[cur ^ 1][0];
#pragma unroll
      for (int j = 0; j < 4; ++j) *(v8h*)(wp + loff[j]) = stage[j];
    }
    cur ^= 1;
  }

  // --- epilogue. C/D layout: lane L, VGPR r -> row M = r + (L>=16 ? 8 : 0),
  // col = L & 15. Bias already folded into acc.
  const int rowBase = m0 + laneHi * 8;
#pragma unroll
  for (int t = 0; t < 2; ++t) {
    const int col = n0 + t * 16 + laneM;
#pragma unroll
    for (int r = 0; r < 8; ++r) {
      const int row = rowBase + r;
      const float fg = sigf(acc[0][t][r]);
      const float ig = sigf(acc[1][t][r]);
      const float cg = tanhf(acc[2][t][r]);
      const float og = sigf(acc[3][t][r]);
      const float cp = c_prev[(size_t)row * HID + col];
      const float cn = fg * cp + ig * cg;
      const float hn = og * tanhf(cn);
      out[(size_t)row * HID + col] = hn;                        // h_t
      out[(size_t)BSZ * HID + (size_t)row * HID + col] = cn;    // c_t
    }
  }
}

// ---------------------------------------------------------------------------
extern "C" void kernel_launch(void* const* d_in, const int* in_sizes, int n_in,
                              void* d_out, int out_size, void* d_ws,
                              size_t ws_size, hipStream_t stream) {
  (void)in_sizes; (void)n_in; (void)out_size; (void)ws_size;
  const float* x      = (const float*)d_in[0];
  const float* h_prev = (const float*)d_in[1];
  const float* c_prev = (const float*)d_in[2];
  const float* W_f = (const float*)d_in[3];
  const float* b_f = (const float*)d_in[4];
  const float* W_i = (const float*)d_in[5];
  const float* b_i = (const float*)d_in[6];
  const float* W_c = (const float*)d_in[7];
  const float* b_c = (const float*)d_in[8];
  const float* W_o = (const float*)d_in[9];
  const float* b_o = (const float*)d_in[10];

  // Workspace layout (needs 32 MB):
  //   [0, 16MB)  : comb f16 [BSZ][KDIM]
  //   [16, 32MB) : Wt   f16 [4][HID][KDIM]
  _Float16* comb = (_Float16*)d_ws;
  _Float16* Wt   = comb + (size_t)BSZ * KDIM;

  // 1) combined = f16(cat(x, h))
  build_combined<<<(BSZ * KDIM / 2) / 256, 256, 0, stream>>>(x, h_prev, comb);

  // 2) transpose+convert weights
  dim3 tb(32, 8);
  dim3 tg(HID / 32, KDIM / 32, 4);
  wt_kernel<<<tg, tb, 0, stream>>>(W_f, W_i, W_c, W_o, Wt);

  // 3) fused WMMA GEMM + LSTM cell
  lstm_gemm<<<(BSZ / 128) * (HID / 32), 256, 0, stream>>>(
      comb, Wt, c_prev, b_f, b_i, b_c, b_o, (float*)d_out);
}
